// SelfAttentionPairDim2_44444321579151
// MI455X (gfx1250) — matrix operand order repaired and checked by reading, not compiled, not run on hardware
//
#include <hip/hip_runtime.h>

typedef __attribute__((ext_vector_type(16))) __bf16        v16bf;
typedef __attribute__((ext_vector_type(8)))  float         v8f;
typedef __attribute__((ext_vector_type(4)))  float         v4f;
typedef __attribute__((ext_vector_type(4)))  unsigned int  v4u;

static constexpr int DD = 64;    // head dim
static constexpr int MM = 256;   // pair dim
static constexpr int SR = 64;    // a-rows per workgroup (strip)
static constexpr int KP = 72;    // padded bf16 stride for D-wide rows (bank-conflict padding)
static constexpr int CP = 40;    // padded bf16 stride for 32-wide c-window rows

// fp32 -> bf16 (round-half-up, 2 VALU ops)
static __device__ __forceinline__ unsigned short f2bf(float f) {
  unsigned int u = __builtin_bit_cast(unsigned int, f) + 0x8000u;
  return (unsigned short)(u >> 16);
}
// pack two fp32 -> two bf16 in one dword: 2x v_add + 1x v_perm_b32
static __device__ __forceinline__ unsigned int pkbf(float a, float b) {
  unsigned int ua = __builtin_bit_cast(unsigned int, a) + 0x8000u;
  unsigned int ub = __builtin_bit_cast(unsigned int, b) + 0x8000u;
  // result bytes: {ub[3],ub[2],ua[3],ua[2]}
  return __builtin_amdgcn_perm(ub, ua, 0x07060302u);
}

// Build a 16-element bf16 fragment from two 8-float runs (global, fp32)
static __device__ __forceinline__ v16bf cvt16(const float* p0, const float* p1) {
  union { v16bf v; v4u q[2]; } t;
  v4f a0 = *(const v4f*)p0;
  v4f a1 = *(const v4f*)(p0 + 4);
  v4f b0 = *(const v4f*)p1;
  v4f b1 = *(const v4f*)(p1 + 4);
  t.q[0][0] = pkbf(a0[0], a0[1]);
  t.q[0][1] = pkbf(a0[2], a0[3]);
  t.q[0][2] = pkbf(a1[0], a1[1]);
  t.q[0][3] = pkbf(a1[2], a1[3]);
  t.q[1][0] = pkbf(b0[0], b0[1]);
  t.q[1][1] = pkbf(b0[2], b0[3]);
  t.q[1][2] = pkbf(b1[0], b1[1]);
  t.q[1][3] = pkbf(b1[2], b1[3]);
  return t.v;
}
// B fragment (32x16) from fp32 row-major source row (column fixed per lane):
// elements j -> k = kk + lh*16 + j
static __device__ __forceinline__ v16bf bfrag_g(const float* row, int kk, int lh) {
  const float* p = row + kk + lh * 16;
  return cvt16(p, p + 8);
}
// Fragments from bf16 LDS (two ds_load_b128 each)
static __device__ __forceinline__ v16bf frag_lds(const unsigned short* p0, const unsigned short* p1) {
  union { v16bf v; v4u q[2]; } t;
  t.q[0] = *(const v4u*)p0;
  t.q[1] = *(const v4u*)p1;
  return t.v;
}
// A fragment (16x32): lane holds row, elements j -> k = lh*16 + j
static __device__ __forceinline__ v16bf afrag_l(const unsigned short* row, int kk, int lh) {
  return frag_lds(row + kk + lh * 16, row + kk + lh * 16 + 8);
}
static __device__ __forceinline__ v16bf bfrag_l(const unsigned short* row, int kk, int lh) {
  const unsigned short* p = row + kk + lh * 16;
  return frag_lds(p, p + 8);
}

static __device__ __forceinline__ v8f wmma_bf16(v16bf a, v16bf b, v8f c) {
  // (neg_a, A, neg_b, B, c_mod, C, reuse_a, reuse_b)
  return __builtin_amdgcn_wmma_f32_16x16x32_bf16(false, a, false, b, (short)0, c, false, false);
}

// Cooperatively stage 32 fp32 rows (64 wide) into bf16 LDS [32][KP].
// Each element is loaded and converted exactly once across the block.
static __device__ __forceinline__ void stageX(unsigned short* Xw, const float* rowbase, int tid) {
  const int r = tid >> 3;          // 32 rows x 8 threads
  const int seg = tid & 7;         // 8 floats per thread
  const float* src = rowbase + (size_t)r * DD + seg * 8;
  v4f f0 = *(const v4f*)src;
  v4f f1 = *(const v4f*)(src + 4);
  v4u pk;
  pk[0] = pkbf(f0[0], f0[1]);
  pk[1] = pkbf(f0[2], f0[3]);
  pk[2] = pkbf(f1[0], f1[1]);
  pk[3] = pkbf(f1[2], f1[3]);
  *(v4u*)&Xw[r * KP + seg * 8] = pk;   // one ds_store_b128
}

__global__ __launch_bounds__(256) void pair_attn_kernel(
    const float* __restrict__ x,     // [B*N, M, D]
    const float* __restrict__ Wq, const float* __restrict__ bq,
    const float* __restrict__ Wk, const float* __restrict__ bk,
    const float* __restrict__ Wv, const float* __restrict__ bv,
    const float* __restrict__ Wo, const float* __restrict__ bo,
    const float* __restrict__ lng, const float* __restrict__ lnb,
    float* __restrict__ out)
{
  __shared__ __align__(16) unsigned short Xw[32 * KP];   // staged bf16 x window [row][e]
  __shared__ __align__(16) unsigned short Kl[SR * KP];   // K strip, row-major [a][d] bf16
  __shared__ __align__(16) unsigned short Qc[32 * KP];   // Q window, row-major [c][d] bf16
  __shared__ __align__(16) unsigned short VTc[DD * CP];  // V window transposed [d][c] bf16
  __shared__ __align__(16) unsigned short Pc[SR * CP];   // relu(S)/8 window [a][c] bf16
  __shared__ __align__(16) unsigned short Cb[SR * KP];   // normalized ctx [a][d] bf16
  __shared__ __align__(16) float          Ybl[SR * KP];  // out-proj result [a][d] fp32
  __shared__ float rowsum[SR];

  const int tid  = threadIdx.x;
  const int wv   = tid >> 5;    // wave id 0..7
  const int lane = tid & 31;
  const int lh   = lane >> 4;   // half-wave
  const int l16  = lane & 15;

  const int bi    = blockIdx.x >> 2;
  const int strip = blockIdx.x & 3;
  const int a0    = strip * SR;
  const float* xb = x   + (size_t)bi * (MM * DD);
  float*       ob = out + (size_t)bi * (MM * DD);

  if (tid < SR) rowsum[tid] = (float)MM * 1e-12f;   // reference adds 1e-12 before the sum

  // per-wave loop-invariant projection state: wave -> (row-subtile, feature-tile)
  const int pct = wv >> 2, pnt = wv & 3;
  const int pd  = pnt * 16 + l16;            // output feature column for this lane
  const v16bf wk0  = bfrag_g(Wk + pd * DD, 0,  lh);
  const v16bf wk32 = bfrag_g(Wk + pd * DD, 32, lh);
  const v16bf wq0  = bfrag_g(Wq + pd * DD, 0,  lh);
  const v16bf wq32 = bfrag_g(Wq + pd * DD, 32, lh);
  const v16bf wv0  = bfrag_g(Wv + pd * DD, 0,  lh);
  const v16bf wv32 = bfrag_g(Wv + pd * DD, 32, lh);
  const float bkd = bk[pd];
  const float bqd = bq[pd];
  const float bvd = bv[pd];

  // ---- phase 0: K strip = x[a0:a0+64] @ Wk^T + bk, via two staged 32-row windows ----
#pragma unroll
  for (int w2 = 0; w2 < 2; ++w2) {
    stageX(Xw, xb + (size_t)(a0 + w2 * 32) * DD, tid);
    __syncthreads();
    {
      const int at = w2 * 2 + pct;
      const unsigned short* arow = Xw + (pct * 16 + l16) * KP;
      v8f acc = {};
      acc = wmma_bf16(afrag_l(arow, 0,  lh), wk0,  acc);
      acc = wmma_bf16(afrag_l(arow, 32, lh), wk32, acc);
#pragma unroll
      for (int r = 0; r < 8; ++r) {
        const int m = at * 16 + r + 8 * lh;
        Kl[m * KP + pd] = f2bf(acc[r] + bkd);
      }
    }
    __syncthreads();
  }

  // ctx accumulators: wave owns output tiles 2*wv and 2*wv+1 (at = t>>2, dt = t&3)
  v8f ctx0 = {}, ctx1 = {};
  // per-lane partial row sums of relu(S)/8, reduced once after the loop
  float psum[8] = {0.f, 0.f, 0.f, 0.f, 0.f, 0.f, 0.f, 0.f};
  const int sat = wv >> 1, sct = wv & 1;     // S-tile coords for this wave

  // ---- phase 1: stream c in 32-wide windows ----
  for (int cc = 0; cc < 8; ++cc) {
    const int c0 = cc * 32;

    // stage x window once (each element converted exactly once)
    stageX(Xw, xb + (size_t)c0 * DD, tid);
    __syncthreads();

    // 1a: Q window (row-major) and V window (transposed); A fragments shared from LDS
    {
      const unsigned short* arow = Xw + (pct * 16 + l16) * KP;
      const v16bf xa0  = afrag_l(arow, 0,  lh);
      const v16bf xa32 = afrag_l(arow, 32, lh);
      v8f accq = {};
      accq = wmma_bf16(xa0,  wq0,  accq);
      accq = wmma_bf16(xa32, wq32, accq);
      v8f accv = {};
      accv = wmma_bf16(xa0,  wv0,  accv);
      accv = wmma_bf16(xa32, wv32, accv);
#pragma unroll
      for (int r = 0; r < 8; ++r) {
        const int m = pct * 16 + r + 8 * lh;
        Qc[m * KP + pd] = f2bf(accq[r] + bqd);
      }
      v4u pk;
#pragma unroll
      for (int q = 0; q < 4; ++q)
        pk[q] = pkbf(accv[2 * q] + bvd, accv[2 * q + 1] + bvd);
      // 8 consecutive c values of column pd -> contiguous 16B in VTc
      *(v4u*)&VTc[pd * CP + pct * 16 + 8 * lh] = pk;
    }
    __syncthreads();

    // 1b: S tile = K(a-tile) . Q(c-tile)^T, scale + relu + partial row sums + bf16 probs
    {
      const unsigned short* arow = Kl + (sat * 16 + l16) * KP;
      const unsigned short* brow = Qc + (sct * 16 + l16) * KP;
      v8f s = {};
      s = wmma_bf16(afrag_l(arow, 0,  lh), bfrag_l(brow, 0,  lh), s);
      s = wmma_bf16(afrag_l(arow, 32, lh), bfrag_l(brow, 32, lh), s);
#pragma unroll
      for (int r = 0; r < 8; ++r) {
        float v = s[r] * 0.125f;             // 1/sqrt(64)
        v = v > 0.f ? v : 0.f;               // relu
        psum[r] += v;                        // defer cross-lane reduction
        const int m = sat * 16 + r + 8 * lh;
        Pc[m * CP + sct * 16 + l16] = f2bf(v);
      }
    }
    __syncthreads();

    // 1c: ctx += P(window) . V(window)   (K = 32 per window)
#pragma unroll
    for (int t = 0; t < 2; ++t) {
      const int tile = 2 * wv + t;
      const int at = tile >> 2, dt = tile & 3;
      const unsigned short* arow = Pc  + (at * 16 + l16) * CP;
      const unsigned short* brow = VTc + (dt * 16 + l16) * CP;
      v8f& acc = t ? ctx1 : ctx0;
      acc = wmma_bf16(afrag_l(arow, 0, lh), bfrag_l(brow, 0, lh), acc);
    }
    __syncthreads();
  }

  // ---- finalize rowsum: one tree reduction per wave, two waves combine via LDS atomics ----
#pragma unroll
  for (int r = 0; r < 8; ++r) {
    float t = psum[r];
    t += __shfl_xor(t, 1, 16);
    t += __shfl_xor(t, 2, 16);
    t += __shfl_xor(t, 4, 16);
    t += __shfl_xor(t, 8, 16);
    const int m = sat * 16 + r + 8 * lh;
    if (l16 == 0) atomicAdd(&rowsum[m], t);   // ds_add_f32
  }
  __syncthreads();

  // ---- phase 2a: normalize ctx by rowsum, stage as bf16 ----
#pragma unroll
  for (int t = 0; t < 2; ++t) {
    const int tile = 2 * wv + t;
    const int at = tile >> 2, dt = tile & 3;
    v8f acc = t ? ctx1 : ctx0;
    const int d = dt * 16 + l16;
#pragma unroll
    for (int r = 0; r < 8; ++r) {
      const int m = at * 16 + r + 8 * lh;
      Cb[m * KP + d] = f2bf(acc[r] / rowsum[m]);
    }
  }
  __syncthreads();

  // ---- phase 2b: out = ctx @ Wo^T + bo ----
#pragma unroll
  for (int t = 0; t < 2; ++t) {
    const int tile = 2 * wv + t;
    const int at = tile >> 2, nt = tile & 3;
    const unsigned short* arow = Cb + (at * 16 + l16) * KP;
    const int d = nt * 16 + l16;
    v8f acc = {};
    acc = wmma_bf16(afrag_l(arow, 0,  lh), bfrag_g(Wo + d * DD, 0,  lh), acc);
    acc = wmma_bf16(afrag_l(arow, 32, lh), bfrag_g(Wo + d * DD, 32, lh), acc);
    const float bias = bo[d];
#pragma unroll
    for (int r = 0; r < 8; ++r) {
      const int m = at * 16 + r + 8 * lh;
      Ybl[m * KP + d] = acc[r] + bias;
    }
  }
  __syncthreads();

  // ---- phase 2c: residual + LayerNorm + store (fp32 throughout) ----
  {
    const int row = tid >> 2;            // 64 rows x 4 threads
    const int d0  = (tid & 3) * 16;
    const int a   = a0 + row;
    const float* xr = xb + (size_t)a * DD + d0;
    float y[16];
    float sum = 0.f, sq = 0.f;
#pragma unroll
    for (int i = 0; i < 16; i += 4) {
      v4f xv = *(const v4f*)(xr + i);
#pragma unroll
      for (int k = 0; k < 4; ++k) {
        float v = Ybl[row * KP + d0 + i + k] + xv[k];
        y[i + k] = v;
        sum += v;
        sq  += v * v;
      }
    }
    sum += __shfl_xor(sum, 1, 4);
    sum += __shfl_xor(sum, 2, 4);
    sq  += __shfl_xor(sq, 1, 4);
    sq  += __shfl_xor(sq, 2, 4);
    const float mu  = sum * (1.f / 64.f);
    const float var = sq * (1.f / 64.f) - mu * mu;
    const float rs  = rsqrtf(var + 1e-5f);
    float* orow = ob + (size_t)a * DD + d0;
#pragma unroll
    for (int i = 0; i < 16; i += 4) {
      v4f g = *(const v4f*)(lng + d0 + i);
      v4f b = *(const v4f*)(lnb + d0 + i);
      v4f o;
#pragma unroll
      for (int k = 0; k < 4; ++k) o[k] = (y[i + k] - mu) * rs * g[k] + b[k];
      *(v4f*)(orow + i) = o;
    }
  }
}

extern "C" void kernel_launch(void* const* d_in, const int* in_sizes, int n_in,
                              void* d_out, int out_size, void* d_ws, size_t ws_size,
                              hipStream_t stream) {
  (void)in_sizes; (void)n_in; (void)out_size; (void)d_ws; (void)ws_size;
  // d_in order: hidden_states(unused), structure_matrix, Wq,bq, Wk,bk, Wv,bv, Wo,bo, ln_g,ln_b
  const float* x   = (const float*)d_in[1];
  const float* Wq  = (const float*)d_in[2];
  const float* bq  = (const float*)d_in[3];
  const float* Wk  = (const float*)d_in[4];
  const float* bk  = (const float*)d_in[5];
  const float* Wv  = (const float*)d_in[6];
  const float* bv  = (const float*)d_in[7];
  const float* Wo  = (const float*)d_in[8];
  const float* bo  = (const float*)d_in[9];
  const float* lng = (const float*)d_in[10];
  const float* lnb = (const float*)d_in[11];
  float* out = (float*)d_out;

  const int blocks = 4 * 256 * (MM / SR);   // B * N * strips = 4096
  pair_attn_kernel<<<blocks, 256, 0, stream>>>(x, Wq, bq, Wk, bk, Wv, bv, Wo, bo,
                                               lng, lnb, out);
}
